// BGEM3Model_16913581211731
// MI455X (gfx1250) — matrix operand order repaired and checked by reading, not compiled, not run on hardware
//
#include <hip/hip_runtime.h>
#include <hip/hip_bf16.h>
#include <stdint.h>

typedef unsigned short u16;
typedef __attribute__((ext_vector_type(16))) __bf16 v16bf;
typedef __attribute__((ext_vector_type(8)))  float  v8f;
typedef __attribute__((ext_vector_type(4)))  unsigned int u32x4;
typedef __attribute__((ext_vector_type(2)))  unsigned int u32x2;
typedef __attribute__((ext_vector_type(4)))  float  f32x4;

#define TEMP 0.02f

__device__ __forceinline__ u16 f2bf(float f) {
  unsigned u = __builtin_bit_cast(unsigned, f);
  u += 0x7FFFu + ((u >> 16) & 1u);               // round-to-nearest-even
  return (u16)(u >> 16);
}
__device__ __forceinline__ float bf2f(u16 h) {
  unsigned u = ((unsigned)h) << 16;
  return __builtin_bit_cast(float, u);
}

union Frag { v16bf v; u32x4 q[2]; };

#define WMMA_BF16(A, B, C) \
  __builtin_amdgcn_wmma_f32_16x16x32_bf16(false, (A), false, (B), (short)0, (C), false, false)

// Async global->LDS copy of 16 bytes (CDNA5 Tensor/Async path, ASYNCcnt-tracked).
__device__ __forceinline__ void async_load_b128(const u16* gaddr, unsigned lds_off) {
  asm volatile("global_load_async_to_lds_b128 %0, %1, off"
               :: "v"(lds_off), "v"(gaddr)
               : "memory");
}

// ---------------------------------------------------------------------------
// Kernel 0: convert colbert_w (f32, [1024][1024] row-major = [dim][h]) to bf16
// ---------------------------------------------------------------------------
__global__ void __launch_bounds__(256) convert_w_kernel(const float* __restrict__ w,
                                                        u16* __restrict__ w_bf) {
  int idx = blockIdx.x * 256 + threadIdx.x;     // one float4 per thread
  f32x4 v = ((const f32x4*)w)[idx];
  u32x2 o;
  o.x = (unsigned)f2bf(v.x) | ((unsigned)f2bf(v.y) << 16);
  o.y = (unsigned)f2bf(v.z) | ((unsigned)f2bf(v.w) << 16);
  ((u32x2*)w_bf)[idx] = o;
}

// ---------------------------------------------------------------------------
// Kernel 1: ColBERT projection. out[b][j][d] = sum_h hidden[b][1+j][h]*W[d][h] + bias[d]
// Row j in [0, L-1) real, row L-1 is zero padding. Raw bf16 written to `col`,
// per-token sum-of-squares written to `norms` (normalization is a second pass).
// Block: 256 threads (8 waves), 16 token rows, full 1024 output dims.
// ---------------------------------------------------------------------------
__global__ void __launch_bounds__(256) colbert_project_kernel(
    const float* __restrict__ hidden, const u16* __restrict__ w_bf,
    const float* __restrict__ bias, u16* __restrict__ col,
    float* __restrict__ norms, int L) {
  __shared__ u16   aT[16 * 1032];         // 16 token rows x 1024 (bf16), padded stride
  __shared__ float redW[8][16][17];       // per-wave sumsq partials [wave][row][lane-col]

  const int tilesPerBatch = L >> 4;
  const int b = blockIdx.x / tilesPerBatch;
  const int t = blockIdx.x % tilesPerBatch;
  const int tid = threadIdx.x;

  // Phase 1: stage 16 token rows as bf16 into LDS (pad rows -> zero)
  for (int it = 0; it < 16; ++it) {
    int idx = tid + it * 256;             // 4096 float4 chunks = 16 rows x 256
    int m  = idx >> 8;
    int kq = idx & 255;
    int j  = t * 16 + m;
    f32x4 v = {0.f, 0.f, 0.f, 0.f};
    if (j < L - 1)
      v = *(const f32x4*)(hidden + (((size_t)b * L + 1 + j) << 10) + kq * 4);
    u32x2 o;
    o.x = (unsigned)f2bf(v.x) | ((unsigned)f2bf(v.y) << 16);
    o.y = (unsigned)f2bf(v.z) | ((unsigned)f2bf(v.w) << 16);
    *(u32x2*)(aT + m * 1032 + kq * 4) = o;
  }
  __syncthreads();

  const int wv = tid >> 5, lane = tid & 31;
  const int m16 = lane & 15, hk = lane >> 4;
  float ps[8] = {0.f, 0.f, 0.f, 0.f, 0.f, 0.f, 0.f, 0.f};
  const size_t outRow0 = ((size_t)b * L + t * 16) << 10;
  const u16* arow = aT + m16 * 1032 + hk * 8;

  // Phase 2: each wave computes 8 output-dim tiles of 16 over full K=1024
  for (int jt = wv; jt < 64; jt += 8) {
    v8f acc = {0.f, 0.f, 0.f, 0.f, 0.f, 0.f, 0.f, 0.f};
    const int d = jt * 16 + m16;                      // lane's output dim (C column)
    const u16* wrow = w_bf + ((size_t)d << 10) + hk * 8;
    for (int kk = 0; kk < 32; ++kk) {
      Frag fa, fb;
      const u16* ap = arow + kk * 32;
      fa.q[0] = *(const u32x4*)ap;                    // K = k0 .. k0+7
      fa.q[1] = *(const u32x4*)(ap + 16);             // K = k0+16 .. k0+23
      const u16* bp = wrow + kk * 32;
      fb.q[0] = *(const u32x4*)bp;                    // K = k0 .. k0+7
      fb.q[1] = *(const u32x4*)(bp + 16);             // K = k0+16 .. k0+23
      acc = WMMA_BF16(fa.v, fb.v, acc);
    }
    const float bv = bias[d];
    for (int r = 0; r < 8; ++r) {
      float val = acc[r] + bv;
      ps[r] += val * val;
      int m = r + 8 * hk;                             // C row
      col[outRow0 + ((size_t)m << 10) + jt * 16 + m16] = f2bf(val);
    }
  }

  // Phase 3: per-token sum-of-squares reduction across waves/lanes
  for (int r = 0; r < 8; ++r) redW[wv][r + 8 * hk][m16] = ps[r];
  __syncthreads();
  if (tid < 16) {
    float s = 0.f;
    for (int w = 0; w < 8; ++w)
      for (int n = 0; n < 16; ++n) s += redW[w][tid][n];
    norms[(size_t)b * L + t * 16 + tid] = s;
  }
}

// ---------------------------------------------------------------------------
// Kernel 2: normalize rows in place: v * mask / max(|mask|*||v||, 1e-12);
// pad rows forced to zero. One block per token row.
// ---------------------------------------------------------------------------
__global__ void __launch_bounds__(256) colbert_normalize_kernel(
    u16* __restrict__ col, const float* __restrict__ norms,
    const float* __restrict__ mask, int L) {
  const int token = blockIdx.x;                 // 0 .. B*L-1
  const int b = token / L, j = token % L;
  const float mv = (j < L - 1) ? mask[b * L + 1 + j] : 0.0f;
  const float s = norms[token];
  const float scale = (mv == 0.f) ? 0.f : mv / fmaxf(fabsf(mv) * sqrtf(s), 1e-12f);
  u32x2* p = (u32x2*)(col + ((size_t)token << 10));  // 256 u32x2 per row
  u32x2 vv = p[threadIdx.x];
  float f0 = bf2f((u16)(vv.x & 0xFFFF)) * scale;
  float f1 = bf2f((u16)(vv.x >> 16)) * scale;
  float f2 = bf2f((u16)(vv.y & 0xFFFF)) * scale;
  float f3 = bf2f((u16)(vv.y >> 16)) * scale;
  u32x2 o;
  o.x = (unsigned)f2bf(f0) | ((unsigned)f2bf(f1) << 16);
  o.y = (unsigned)f2bf(f2) | ((unsigned)f2bf(f3) << 16);
  p[threadIdx.x] = o;
}

// ---------------------------------------------------------------------------
// Kernel 3: maxsim. One block per (qb, pb). scores[i][j] = q_col[i] . p_col[j];
// colbert = sum_i max_j scores / count / TEMP. 8 waves x i-tiles; p_col staged
// in LDS in double-buffered 64x128 K-chunks via ASYNC global->LDS loads
// (overlapped with WMMA on the other buffer); 4 accumulators per A-frag build.
// ---------------------------------------------------------------------------
__global__ void __launch_bounds__(256) maxsim_kernel(
    const u16* __restrict__ qcol, const u16* __restrict__ pcol,
    const float* __restrict__ q_mask, float* __restrict__ out) {
  __shared__ u16   bT[2][64 * 136];       // double-buffered: 64 p rows x 128 K-halves
  __shared__ float redm[128 * 17];
  __shared__ float rsum[128];

  const int pb = blockIdx.x, qb = blockIdx.y;
  const int tid = threadIdx.x, wv = tid >> 5, lane = tid & 31;
  const int m16 = lane & 15, hk = lane >> 4;

  // this thread's 4 staging slots within a 64x128 chunk
  const int sr[4] = { (tid + 0) >> 4, (tid + 256) >> 4, (tid + 512) >> 4, (tid + 768) >> 4 };
  const int sc = (tid & 15) * 8;          // column (halves) within row, same for all 4

  // issue the 4 async loads for chunk c (c = jtg*8 + kc) into buffer (c&1)
  auto issue_chunk = [&](int c) {
    const int jtg2 = c >> 3, kc2 = c & 7;
    const u16* srcBase = pcol + (((size_t)pb * 512 + jtg2 * 64) << 10) + kc2 * 128 + sc;
    u16* dstBase = &bT[c & 1][sc];
#pragma unroll
    for (int it = 0; it < 4; ++it) {
      const u16* g = srcBase + ((size_t)sr[it] << 10);
      unsigned lds_off = (unsigned)(uintptr_t)(dstBase + sr[it] * 136);
      async_load_b128(g, lds_off);        // ASYNCcnt += 1 (4 per wave per chunk)
    }
  };

  float runmax[8];
  for (int r = 0; r < 8; ++r) runmax[r] = -3.4e38f;

  const u16* qrow = qcol + (((size_t)qb * 128 + wv * 16 + m16) << 10) + hk * 8;

  issue_chunk(0);                          // prime the pipeline
  for (int jtg = 0; jtg < 8; ++jtg) {      // 8 groups of 64 j each
    v8f a0 = {0.f,0.f,0.f,0.f,0.f,0.f,0.f,0.f};
    v8f a1 = a0, a2 = a0, a3 = a0;
    for (int kc = 0; kc < 8; ++kc) {       // K chunks of 128
      const int c = jtg * 8 + kc;
      if (c + 1 < 64) {
        issue_chunk(c + 1);                // prefetch next chunk into other buffer
        asm volatile("s_wait_asynccnt 0x4" ::: "memory");  // chunk c landed
      } else {
        asm volatile("s_wait_asynccnt 0x0" ::: "memory");
      }
      __syncthreads();                     // all waves' async data visible
      const u16* buf = bT[c & 1];
      for (int kk = 0; kk < 4; ++kk) {
        Frag fa;
        const u16* ap = qrow + kc * 128 + kk * 32;
        fa.q[0] = *(const u32x4*)ap;
        fa.q[1] = *(const u32x4*)(ap + 16);
        const u16* bbase = buf + m16 * 136 + kk * 32 + hk * 8;
        Frag fb;
        fb.q[0] = *(const u32x4*)(bbase);
        fb.q[1] = *(const u32x4*)(bbase + 16);
        a0 = WMMA_BF16(fa.v, fb.v, a0);
        fb.q[0] = *(const u32x4*)(bbase + 16 * 136);
        fb.q[1] = *(const u32x4*)(bbase + 16 * 136 + 16);
        a1 = WMMA_BF16(fa.v, fb.v, a1);
        fb.q[0] = *(const u32x4*)(bbase + 32 * 136);
        fb.q[1] = *(const u32x4*)(bbase + 32 * 136 + 16);
        a2 = WMMA_BF16(fa.v, fb.v, a2);
        fb.q[0] = *(const u32x4*)(bbase + 48 * 136);
        fb.q[1] = *(const u32x4*)(bbase + 48 * 136 + 16);
        a3 = WMMA_BF16(fa.v, fb.v, a3);
      }
      __syncthreads();                     // done reading buf before it is re-filled
    }
    // running max over real j columns only (lane's column j = jtg*64 + t*16 + m16)
    const int jb = jtg * 64 + m16;
    if (jb      <= 510) for (int r = 0; r < 8; ++r) runmax[r] = fmaxf(runmax[r], a0[r]);
    if (jb + 16 <= 510) for (int r = 0; r < 8; ++r) runmax[r] = fmaxf(runmax[r], a1[r]);
    if (jb + 32 <= 510) for (int r = 0; r < 8; ++r) runmax[r] = fmaxf(runmax[r], a2[r]);
    if (jb + 48 <= 510) for (int r = 0; r < 8; ++r) runmax[r] = fmaxf(runmax[r], a3[r]);
  }

  // cross-lane max per i row, then sum over i
  for (int r = 0; r < 8; ++r)
    redm[(wv * 16 + r + 8 * hk) * 17 + m16] = runmax[r];
  __syncthreads();
  if (tid < 128) {
    float mx = redm[tid * 17];
    for (int c = 1; c < 16; ++c) mx = fmaxf(mx, redm[tid * 17 + c]);
    rsum[tid] = mx;                        // row 127 is the zero pad row -> contributes 0
  }
  __syncthreads();
  for (int s = 64; s > 0; s >>= 1) {
    if (tid < s) rsum[tid] += rsum[tid + s];
    __syncthreads();
  }
  if (tid == 0) {
    float denom = 0.f;
    for (int l = 1; l < 128; ++l) denom += q_mask[qb * 128 + l];
    out[qb * 64 + pb] = rsum[0] / denom / TEMP;
  }
}

// ---------------------------------------------------------------------------
// Kernel 4: dense scores (CLS pooling + cosine / TEMP). One block per (qb,pb).
// ---------------------------------------------------------------------------
__global__ void __launch_bounds__(256) dense_kernel(
    const float* __restrict__ q_hidden, const float* __restrict__ p_hidden,
    float* __restrict__ out) {
  __shared__ float rd[3][8];
  const int pb = blockIdx.x, qb = blockIdx.y, tid = threadIdx.x;
  const float* qr = q_hidden + (((size_t)qb * 128) << 10);
  const float* pr = p_hidden + (((size_t)pb * 512) << 10);
  float dot = 0.f, qs = 0.f, psn = 0.f;
  for (int k = tid; k < 1024; k += 256) {
    float a = qr[k], c = pr[k];
    dot += a * c; qs += a * a; psn += c * c;
  }
  for (int off = 16; off; off >>= 1) {
    dot += __shfl_xor(dot, off);
    qs  += __shfl_xor(qs, off);
    psn += __shfl_xor(psn, off);
  }
  if ((tid & 31) == 0) { rd[0][tid >> 5] = dot; rd[1][tid >> 5] = qs; rd[2][tid >> 5] = psn; }
  __syncthreads();
  if (tid == 0) {
    float D = 0.f, Q = 0.f, P = 0.f;
    for (int w = 0; w < 8; ++w) { D += rd[0][w]; Q += rd[1][w]; P += rd[2][w]; }
    out[qb * 64 + pb] = D / (fmaxf(sqrtf(Q), 1e-12f) * fmaxf(sqrtf(P), 1e-12f)) / TEMP;
  }
}

// ---------------------------------------------------------------------------
// Kernel 5a: per-token sparse weights tw = relu(hidden . sparse_w + sparse_b)
// One wave per token (8 waves/block). Token ids flat: q first (1024), p next (32768).
// ---------------------------------------------------------------------------
__global__ void __launch_bounds__(256) sparse_tw_kernel(
    const float* __restrict__ q_hidden, const float* __restrict__ p_hidden,
    const float* __restrict__ sparse_w, const float* __restrict__ sparse_b,
    float* __restrict__ qtw, float* __restrict__ ptw) {
  const int g = blockIdx.x * 8 + (threadIdx.x >> 5);
  const int lane = threadIdx.x & 31;
  const float* row;
  float* dst;
  if (g < 1024) { row = q_hidden + ((size_t)g << 10); dst = qtw + g; }
  else          { int gp = g - 1024; row = p_hidden + ((size_t)gp << 10); dst = ptw + gp; }
  float s = 0.f;
  for (int k = lane; k < 1024; k += 32) s += row[k] * sparse_w[k];
  for (int off = 16; off; off >>= 1) s += __shfl_xor(s, off);
  if (lane == 0) *dst = fmaxf(s + sparse_b[0], 0.0f);
}

// ---------------------------------------------------------------------------
// Kernel 5b: sparse scores by id matching with scatter-max semantics.
// One block per (qb, pb). Ownership dedups duplicate ids within q (argmax).
// Ids 0..3 (cls/pad/eos/unk) excluded per reference.
// ---------------------------------------------------------------------------
__global__ void __launch_bounds__(256) sparse_score_kernel(
    const int* __restrict__ q_ids, const int* __restrict__ p_ids,
    const float* __restrict__ qtw, const float* __restrict__ ptw,
    float* __restrict__ out) {
  __shared__ int   qid_s[128];
  __shared__ float qtw_s[128];
  __shared__ int   pid_s[512];
  __shared__ float ptw_s[512];
  __shared__ float red[128];
  const int pb = blockIdx.x, qb = blockIdx.y, tid = threadIdx.x;
  if (tid < 128) { qid_s[tid] = q_ids[qb * 128 + tid]; qtw_s[tid] = qtw[qb * 128 + tid]; }
  for (int j = tid; j < 512; j += 256) { pid_s[j] = p_ids[pb * 512 + j]; ptw_s[j] = ptw[pb * 512 + j]; }
  __syncthreads();
  float c = 0.f;
  if (tid < 128) {
    const int v = qid_s[tid];
    const float w = qtw_s[tid];
    if (v > 3 && w > 0.f) {
      bool owner = true;
      for (int i2 = 0; i2 < 128; ++i2) {
        if (i2 == tid || qid_s[i2] != v) continue;
        float w2 = qtw_s[i2];
        if (w2 > w || (w2 == w && i2 < tid)) { owner = false; break; }
      }
      if (owner) {
        float pm = 0.f;
        for (int j = 0; j < 512; ++j)
          if (pid_s[j] == v) pm = fmaxf(pm, ptw_s[j]);
        c = w * pm;
      }
    }
    red[tid] = c;
  }
  __syncthreads();
  for (int s = 64; s > 0; s >>= 1) {
    if (tid < s) red[tid] += red[tid + s];
    __syncthreads();
  }
  if (tid == 0) out[qb * 64 + pb] = red[0] / TEMP;
}

// ---------------------------------------------------------------------------
extern "C" void kernel_launch(void* const* d_in, const int* in_sizes, int n_in,
                              void* d_out, int out_size, void* d_ws, size_t ws_size,
                              hipStream_t stream) {
  const float* q_hidden  = (const float*)d_in[0];   // [8,128,1024]
  const float* p_hidden  = (const float*)d_in[1];   // [64,512,1024]
  const float* q_mask    = (const float*)d_in[2];   // [8,128]
  const float* p_mask    = (const float*)d_in[3];   // [64,512]
  const int*   q_ids     = (const int*)d_in[4];     // [8,128]
  const int*   p_ids     = (const int*)d_in[5];     // [64,512]
  const float* colbert_w = (const float*)d_in[6];   // [1024,1024]
  const float* colbert_b = (const float*)d_in[7];   // [1024]
  const float* sparse_w  = (const float*)d_in[8];   // [1024]
  const float* sparse_b  = (const float*)d_in[9];   // [1]
  float* out = (float*)d_out;                       // dense[512] | sparse[512] | colbert[512]

  // Workspace layout (~71.5 MB)
  char* ws = (char*)d_ws;
  size_t off = 0;
  u16* w_bf = (u16*)(ws + off); off += (size_t)1024 * 1024 * 2;          // 2 MB
  u16* qcol = (u16*)(ws + off); off += (size_t)8 * 128 * 1024 * 2;       // 2 MB
  u16* pcol = (u16*)(ws + off); off += (size_t)64 * 512 * 1024 * 2;      // 64 MB
  float* qnorm = (float*)(ws + off); off += (size_t)8 * 128 * 4;
  float* pnorm = (float*)(ws + off); off += (size_t)64 * 512 * 4;
  float* qtw   = (float*)(ws + off); off += (size_t)8 * 128 * 4;
  float* ptw   = (float*)(ws + off); off += (size_t)64 * 512 * 4;
  (void)in_sizes; (void)n_in; (void)out_size; (void)ws_size;

  convert_w_kernel<<<1024, 256, 0, stream>>>(colbert_w, w_bf);

  colbert_project_kernel<<<8 * 8, 256, 0, stream>>>(
      q_hidden, w_bf, colbert_b, qcol, qnorm, 128);
  colbert_project_kernel<<<64 * 32, 256, 0, stream>>>(
      p_hidden, w_bf, colbert_b, pcol, pnorm, 512);

  colbert_normalize_kernel<<<8 * 128, 256, 0, stream>>>(qcol, qnorm, q_mask, 128);
  colbert_normalize_kernel<<<64 * 512, 256, 0, stream>>>(pcol, pnorm, p_mask, 512);

  maxsim_kernel<<<dim3(64, 8), 256, 0, stream>>>(qcol, pcol, q_mask, out + 1024);

  dense_kernel<<<dim3(64, 8), 256, 0, stream>>>(q_hidden, p_hidden, out);

  sparse_tw_kernel<<<(1024 + 64 * 512) / 8, 256, 0, stream>>>(
      q_hidden, p_hidden, sparse_w, sparse_b, qtw, ptw);
  sparse_score_kernel<<<dim3(64, 8), 256, 0, stream>>>(q_ids, p_ids, qtw, ptw, out + 512);
}
